// MultiHeadAttention_25933012533471
// MI455X (gfx1250) — matrix operand order repaired and checked by reading, not compiled, not run on hardware
//
#include <hip/hip_runtime.h>

// Problem constants (reference: B=4, S=2048, D=512, H=8, DH=64)
static constexpr int Bc = 4, Sc = 2048, Dc = 512, Hc = 8, DHc = 64;

typedef _Float16 h2_t  __attribute__((ext_vector_type(2)));
typedef _Float16 h8_t  __attribute__((ext_vector_type(8)));
typedef _Float16 v16h  __attribute__((ext_vector_type(16)));
typedef float    v8f   __attribute__((ext_vector_type(8)));

union Frag16 {
  v16h v;
  h2_t p[8];
  h8_t q[2];
};

// ---------------------------------------------------------------------------
// CDNA5 async global->LDS staging (ASYNCcnt-tracked), with sync fallback.
// Probe round 2 revealed the builtin's signature: param0 is
// "int __attribute__((vector_size(16))) __device__ *"  (addrspace(1), non-const),
// param1 is the LDS (addrspace(3)) pointer, then imm offset + imm cpol.
// ---------------------------------------------------------------------------
#if __has_builtin(__builtin_amdgcn_global_load_async_to_lds_b128)
#define HAVE_ASYNC_LDS 1
#else
#define HAVE_ASYNC_LDS 0
#endif

typedef int i4_vec __attribute__((vector_size(4 * sizeof(int))));
typedef __attribute__((address_space(1))) i4_vec* glb_i4_p;
typedef __attribute__((address_space(3))) i4_vec* lds_i4_p;

// Copy 16 bytes global -> LDS. Async path: no VGPR data movement, per-lane
// addresses, completion via ASYNCcnt.
__device__ __forceinline__ void stage16(_Float16* dstLds, const _Float16* srcGlb) {
#if HAVE_ASYNC_LDS
  __builtin_amdgcn_global_load_async_to_lds_b128(
      (glb_i4_p)srcGlb, (lds_i4_p)dstLds, 0, 0);
#else
  *(h8_t*)dstLds = *(const h8_t*)srcGlb;
#endif
}

__device__ __forceinline__ void wait_async() {
#if HAVE_ASYNC_LDS
#if __has_builtin(__builtin_amdgcn_s_wait_asynccnt)
  __builtin_amdgcn_s_wait_asynccnt(0);
#else
  asm volatile("s_wait_asynccnt 0x0" ::: "memory");
#endif
#endif
}

// D = A(16x32 f16) * B(32x16 f16) + C(16x16 f32)
__device__ __forceinline__ v8f wmma_f16(v16h a, v16h b, v8f c) {
  return __builtin_amdgcn_wmma_f32_16x16x32_f16(
      /*neg_a=*/false, a, /*neg_b=*/false, b,
      /*c_mod=*/(short)0, c, /*reuse_a=*/false, /*reuse_b=*/false);
}

// A-fragment (16x32) from LDS, row-major tile with stride `ld` halfs.
// ISA layout: lanes 0-15 row=lane, K {0..7, 16..23}; lanes 16-31 row=lane-16,
// K {8..15, 24..31}; two f16 per VGPR.
__device__ __forceinline__ v16h load_a_lds(const _Float16* base, int ld, int lane) {
  Frag16 f;
  const int row = lane & 15;
  const int kb  = (lane < 16) ? 0 : 8;
  const _Float16* r = base + row * ld + kb;
#pragma unroll
  for (int j = 0; j < 4; ++j) f.p[j]     = *(const h2_t*)(r + 2 * j);
#pragma unroll
  for (int j = 0; j < 4; ++j) f.p[4 + j] = *(const h2_t*)(r + 16 + 2 * j);
  return f.v;
}

// A-fragment (16x32) directly from global f16 rows of stride `ld` halfs.
__device__ __forceinline__ v16h load_a_global(const _Float16* rowbase, int ld, int lane) {
  Frag16 f;
  const int row = lane & 15;
  const int kb  = (lane < 16) ? 0 : 8;
  const _Float16* r = rowbase + (size_t)row * ld + kb;
  f.q[0] = *(const h8_t*)(r);       // K kb..kb+7
  f.q[1] = *(const h8_t*)(r + 16);  // K kb+16..kb+23
  return f.v;
}

// B-fragment (32x16) from an LDS tile holding B^T (row = column n, contiguous k).
// Layout: lane = column (n), lanes 0-15 K {0..7, 16..23}, lanes 16-31 K {8..15, 24..31}.
__device__ __forceinline__ v16h load_b_lds(const _Float16* base, int ld, int lane,
                                           int nOff, int kOff) {
  Frag16 f;
  const int n  = (lane & 15) + nOff;
  const int kb = ((lane < 16) ? 0 : 8) + kOff;
  const _Float16* r = base + n * ld + kb;
  f.q[0] = *(const h8_t*)(r);
  f.q[1] = *(const h8_t*)(r + 16);
  return f.v;
}

// ---------------------------------------------------------------------------
// Kernel 1: fused Q/K/V projection. C[8192x512] = X[8192x512] * W[512x512]
// (f32 -> f16 on the fly, f32 accumulate in WMMA). z selects Q/K/V.
// Q scaled by 1/sqrt(DH)=0.125 at store; V stored transposed [B,H,DH,S].
// ---------------------------------------------------------------------------
__global__ __launch_bounds__(256)
void qkv_proj_kernel(const float* __restrict__ Xq, const float* __restrict__ Xk,
                     const float* __restrict__ Xv,
                     const float* __restrict__ Wq, const float* __restrict__ Wk,
                     const float* __restrict__ Wv,
                     _Float16* __restrict__ Qh, _Float16* __restrict__ Kh,
                     _Float16* __restrict__ Vt) {
  constexpr int BM = 128, BN = 64, BK = 32, LDA = 40, LDB = 40;
  __shared__ _Float16 sA[BM * LDA];   // X tile  [row][k]
  __shared__ _Float16 sB[BN * LDB];   // W^T tile [n][k]

  const int mat = blockIdx.z;
  const float* X = (mat == 0) ? Xq : (mat == 1) ? Xk : Xv;
  const float* W = (mat == 0) ? Wq : (mat == 1) ? Wk : Wv;
  const int m0 = blockIdx.x * BM;
  const int n0 = blockIdx.y * BN;
  const int t = threadIdx.x;
  const int lane = t & 31;
  const int wave = t >> 5;

  const int arow = t >> 1;            // A staging: 0..127
  const int acb  = (t & 1) * 16;      // 0 or 16
  const int wn   = t & 63;            // W staging column
  const int wkk  = (t >> 6) * 8;      // W staging k-offset

  v8f acc[4] = {};

  for (int k0 = 0; k0 < Dc; k0 += BK) {
    {  // stage A tile, f32 -> f16
      const float* src = X + (size_t)(m0 + arow) * Dc + k0 + acb;
      _Float16* dst = sA + arow * LDA + acb;
#pragma unroll
      for (int i = 0; i < 16; ++i) dst[i] = (_Float16)src[i];
    }
    {  // stage W^T tile: sB[n][k] = W[k0+k][n0+n]
#pragma unroll
      for (int i = 0; i < 8; ++i)
        sB[wn * LDB + wkk + i] = (_Float16)W[(size_t)(k0 + wkk + i) * Dc + n0 + wn];
    }
    if (k0 + BK < Dc) {  // prefetch next k-tile (global_prefetch_b8)
      __builtin_prefetch(X + (size_t)(m0 + arow) * Dc + (k0 + BK) + acb, 0, 0);
      __builtin_prefetch(W + (size_t)(k0 + BK + wkk) * Dc + n0 + wn, 0, 0);
    }
    __syncthreads();
    const v16h a = load_a_lds(sA + wave * 16 * LDA, LDA, lane);
#pragma unroll
    for (int s = 0; s < 4; ++s) {
      const v16h b = load_b_lds(sB, LDB, lane, s * 16, 0);
      acc[s] = wmma_f16(a, b, acc[s]);
    }
    __syncthreads();
  }

  // C layout: VGPR j, lanes 0-15 -> (M=j, N=lane); lanes 16-31 -> (M=j+8, N=lane-16)
  const int prow = (lane >> 4) << 3;
  const int pcol = lane & 15;
#pragma unroll
  for (int s = 0; s < 4; ++s) {
#pragma unroll
    for (int j = 0; j < 8; ++j) {
      const int row = m0 + wave * 16 + prow + j;   // flat B*S row
      const int col = n0 + s * 16 + pcol;          // model dim
      const int bb = row >> 11, ss = row & 2047;
      const int hh = col >> 6,  dh = col & 63;
      const float v = acc[s][j];
      if (mat == 0)
        Qh[(((size_t)bb * Hc + hh) * Sc + ss) * DHc + dh] = (_Float16)(v * 0.125f);
      else if (mat == 1)
        Kh[(((size_t)bb * Hc + hh) * Sc + ss) * DHc + dh] = (_Float16)v;
      else
        Vt[(((size_t)bb * Hc + hh) * DHc + dh) * Sc + ss] = (_Float16)v;
    }
  }
}

// ---------------------------------------------------------------------------
// Kernel 2: flash attention per (b,h). Block = 256 thr = 8 waves, each wave
// owns 16 query rows (block tile 128). Loop over 64 key-blocks of 32 with
// double-buffered async K/V staging: tile i+1 streams into the other LDS
// buffer (cache -> LDS, no VGPRs) while WMMAs consume tile i.
// ---------------------------------------------------------------------------
__global__ __launch_bounds__(256)
void flash_attn_kernel(const _Float16* __restrict__ Qh,
                       const _Float16* __restrict__ Kh,
                       const _Float16* __restrict__ Vt,
                       _Float16* __restrict__ AttnOut) {
  constexpr int LDK = 72;       // 32 keys x 64 dh (+8 pad): 144B rows (16B mult)
  constexpr int LDV = 40;       // 64 dh  x 32 keys (+8 pad): 80B rows
  constexpr int LDP = 34;       // per-wave P scratch 16x32 (+2 pad)
  constexpr int KBUF = 32 * LDK;
  constexpr int VBUF = 64 * LDV;
  __shared__ _Float16 sK[2 * KBUF];     // [buf][key][dh]  == B^T for Q*K^T
  __shared__ _Float16 sV[2 * VBUF];     // [buf][dh][key]  == B^T for P*V
  __shared__ _Float16 sP[8][16 * LDP];  // per-wave P tile (C->A relayout)

  const int bh = blockIdx.y;            // 0..B*H-1
  const int t = threadIdx.x;
  const int lane = t & 31;
  const int wave = t >> 5;
  const int q0 = blockIdx.x * 128 + wave * 16;  // query row within S

  const _Float16* Qbase = Qh + ((size_t)bh * Sc + q0) * DHc;
  const v16h qf0 = load_a_global(Qbase, DHc, lane);        // k = 0..31 of DH
  const v16h qf1 = load_a_global(Qbase + 32, DHc, lane);   // k = 32..63

  // Per-thread staging slices (16B each for K and V)
  const int skey = t >> 3, sdb = (t & 7) * 8;   // K: 32 keys x 8 chunks
  const int sdh  = t >> 2, skb = (t & 3) * 8;   // V: 64 dh x 4 chunks

  auto stageKV = [&](int k0, int buf) {
    stage16(sK + buf * KBUF + skey * LDK + sdb,
            Kh + ((size_t)bh * Sc + k0 + skey) * DHc + sdb);
    stage16(sV + buf * VBUF + sdh * LDV + skb,
            Vt + ((size_t)bh * DHc + sdh) * Sc + k0 + skb);
  };

  v8f o[4] = {};
  float m[8], l[8];
#pragma unroll
  for (int j = 0; j < 8; ++j) { m[j] = -1e30f; l[j] = 0.0f; }

  stageKV(0, 0);  // prologue: tile 0 -> buffer 0

  for (int k0 = 0; k0 < Sc; k0 += 32) {
    const int buf = (k0 >> 5) & 1;
    wait_async();        // my staging ops (incl. this tile's) complete
    __syncthreads();     // everyone's complete -> buf fully populated
    if (k0 + 32 < Sc) stageKV(k0 + 32, buf ^ 1);  // overlap next tile

    const _Float16* cK = sK + buf * KBUF;
    const _Float16* cV = sV + buf * VBUF;

    // S = Q * K^T : 16 rows x 32 keys (two 16-col subtiles), DH=64 -> 2 k-steps
    v8f s0 = {}, s1 = {};
    s0 = wmma_f16(qf0, load_b_lds(cK, LDK, lane, 0, 0),  s0);
    s0 = wmma_f16(qf1, load_b_lds(cK, LDK, lane, 0, 32), s0);
    s1 = wmma_f16(qf0, load_b_lds(cK, LDK, lane, 16, 0),  s1);
    s1 = wmma_f16(qf1, load_b_lds(cK, LDK, lane, 16, 32), s1);

    // Online softmax: row r = prow+j lives across one 16-lane half.
    float scale[8];
#pragma unroll
    for (int j = 0; j < 8; ++j) {
      float mx = fmaxf(s0[j], s1[j]);
#pragma unroll
      for (int off = 1; off < 16; off <<= 1) mx = fmaxf(mx, __shfl_xor(mx, off));
      const float mnew = fmaxf(m[j], mx);
      scale[j] = __expf(m[j] - mnew);
      s0[j] = __expf(s0[j] - mnew);
      s1[j] = __expf(s1[j] - mnew);
      float rs = s0[j] + s1[j];
#pragma unroll
      for (int off = 1; off < 16; off <<= 1) rs += __shfl_xor(rs, off);
      l[j] = l[j] * scale[j] + rs;
      m[j] = mnew;
    }
#pragma unroll
    for (int s = 0; s < 4; ++s)
#pragma unroll
      for (int j = 0; j < 8; ++j) o[s][j] *= scale[j];

    // P (f32, C-layout) -> f16 via per-wave LDS -> A-layout fragment
    {
      const int prow = (lane >> 4) << 3;
      const int pcol = lane & 15;
      _Float16* pw = sP[wave];
#pragma unroll
      for (int j = 0; j < 8; ++j) {
        pw[(prow + j) * LDP + pcol]      = (_Float16)s0[j];
        pw[(prow + j) * LDP + pcol + 16] = (_Float16)s1[j];
      }
    }
    const v16h pf = load_a_lds(sP[wave], LDP, lane);  // same-wave DS: in order

    // O += P * V : k = 32 keys = exactly one WMMA k-step, 4 dh subtiles
#pragma unroll
    for (int s = 0; s < 4; ++s)
      o[s] = wmma_f16(pf, load_b_lds(cV, LDV, lane, s * 16, 0), o[s]);
    // no trailing barrier: next overwrite of this buffer is issued only after
    // the NEXT iteration's top barrier, which orders it after these reads.
  }

  // Normalize by row sums, merge heads: AttnOut[b, s, h*64+dh] (f16)
  const int prow = (lane >> 4) << 3;
  const int pcol = lane & 15;
  const int bb = bh >> 3, hh = bh & 7;
#pragma unroll
  for (int s = 0; s < 4; ++s) {
#pragma unroll
    for (int j = 0; j < 8; ++j) {
      const size_t row = (size_t)bb * Sc + (q0 + prow + j);
      const int col = hh * DHc + s * 16 + pcol;
      AttnOut[row * Dc + col] = (_Float16)(o[s][j] / l[j]);
    }
  }
}

// ---------------------------------------------------------------------------
// Kernel 3: output projection. out[8192x512] f32 = AttnOut(f16) * W_out(f32->f16)
// A tile is a pure f16 copy -> async global->LDS staging.
// ---------------------------------------------------------------------------
__global__ __launch_bounds__(256)
void out_proj_kernel(const _Float16* __restrict__ Ain,
                     const float* __restrict__ W, float* __restrict__ out) {
  constexpr int BM = 128, BN = 64, BK = 32, LDA = 40, LDB = 40;
  __shared__ _Float16 sA[BM * LDA];
  __shared__ _Float16 sB[BN * LDB];
  const int m0 = blockIdx.x * BM;
  const int n0 = blockIdx.y * BN;
  const int t = threadIdx.x;
  const int lane = t & 31;
  const int wave = t >> 5;

  const int arow = t >> 1;
  const int acb  = (t & 1) * 16;
  const int wn   = t & 63;
  const int wkk  = (t >> 6) * 8;

  v8f acc[4] = {};

  for (int k0 = 0; k0 < Dc; k0 += BK) {
    {  // stage A tile (f16): two async 16B copies per thread
      const _Float16* src = Ain + (size_t)(m0 + arow) * Dc + k0 + acb;
      _Float16* dst = sA + arow * LDA + acb;
      stage16(dst, src);
      stage16(dst + 8, src + 8);
    }
    {  // stage W_out^T tile (f32 -> f16)
#pragma unroll
      for (int i = 0; i < 8; ++i)
        sB[wn * LDB + wkk + i] = (_Float16)W[(size_t)(k0 + wkk + i) * Dc + n0 + wn];
    }
    if (k0 + BK < Dc) {
      __builtin_prefetch(Ain + (size_t)(m0 + arow) * Dc + (k0 + BK) + acb, 0, 0);
      __builtin_prefetch(W + (size_t)(k0 + BK + wkk) * Dc + n0 + wn, 0, 0);
    }
    wait_async();
    __syncthreads();
    const v16h a = load_a_lds(sA + wave * 16 * LDA, LDA, lane);
#pragma unroll
    for (int s = 0; s < 4; ++s) {
      const v16h b = load_b_lds(sB, LDB, lane, s * 16, 0);
      acc[s] = wmma_f16(a, b, acc[s]);
    }
    __syncthreads();
  }

  const int prow = (lane >> 4) << 3;
  const int pcol = lane & 15;
#pragma unroll
  for (int s = 0; s < 4; ++s)
#pragma unroll
    for (int j = 0; j < 8; ++j) {
      const size_t row = (size_t)(m0 + wave * 16 + prow + j);
      out[row * Dc + n0 + s * 16 + pcol] = acc[s][j];
    }
}

// ---------------------------------------------------------------------------
extern "C" void kernel_launch(void* const* d_in, const int* in_sizes, int n_in,
                              void* d_out, int out_size, void* d_ws, size_t ws_size,
                              hipStream_t stream) {
  (void)in_sizes; (void)n_in; (void)out_size; (void)ws_size;
  const float* Xq = (const float*)d_in[0];
  const float* Xk = (const float*)d_in[1];
  const float* Xv = (const float*)d_in[2];
  const float* Wq = (const float*)d_in[3];
  const float* Wk = (const float*)d_in[4];
  const float* Wv = (const float*)d_in[5];
  const float* Wo = (const float*)d_in[6];

  // Workspace layout (f16): Qh | Kh | Vt | AttnOut, each B*H*S*DH = 4 Mi halfs
  const size_t nEl = (size_t)Bc * Hc * Sc * DHc;  // 4,194,304
  _Float16* ws = (_Float16*)d_ws;
  _Float16* Qh = ws;
  _Float16* Kh = ws + nEl;
  _Float16* Vt = ws + 2 * nEl;
  _Float16* Ao = ws + 3 * nEl;

  const dim3 blk(256);
  // 8192/128 = 64 row tiles, 512/64 = 8 col tiles, z = {Q,K,V}
  qkv_proj_kernel<<<dim3(64, 8, 3), blk, 0, stream>>>(Xq, Xk, Xv, Wq, Wk, Wv,
                                                      Qh, Kh, Vt);
  // 2048/128 = 16 query tiles, 32 = B*H
  flash_attn_kernel<<<dim3(16, 32), blk, 0, stream>>>(Qh, Kh, Vt, Ao);
  // final projection to f32 output
  out_proj_kernel<<<dim3(64, 8), blk, 0, stream>>>(Ao, Wo, (float*)d_out);
}
